// SpVecsUnetGcn_18348100288871
// MI455X (gfx1250) — matrix operand order repaired and checked by reading, not compiled, not run on hardware
//
#include <hip/hip_runtime.h>

typedef __attribute__((ext_vector_type(16))) _Float16 v16h;
typedef __attribute__((ext_vector_type(8)))  _Float16 v8h;
typedef __attribute__((ext_vector_type(2)))  _Float16 h2;
typedef __attribute__((ext_vector_type(8)))  float    v8f;

#define P_PIX   (1024*1024)
#define H_IMG   1024
#define N_NODES 16384
#define N_EDGES 65536
#define C_EMB   16
#define C3_DIM  48

__device__ __forceinline__ v8f wmma32(v16h a, v16h b, v8f c) {
  // D = A(16x32 f16) * B(32x16 f16) + C(16x16 f32)
  return __builtin_amdgcn_wmma_f32_16x16x32_f16(false, a, false, b, (short)0, c,
                                                false, false);
}

// A-matrix 16x32 f16 per-lane K pair base (contiguous: k = 16*lh + 2*i). Always even.
__device__ __forceinline__ int kbA(int i, int lh) {
  return 2*i + lh*16;
}

// 16 contiguous f32 -> v16h (four float4 loads)
__device__ __forceinline__ v16h load16f(const float* __restrict__ p) {
  const float4* q = (const float4*)p;
  float4 f0 = q[0], f1 = q[1], f2 = q[2], f3 = q[3];
  v16h r;
  r[0]=(_Float16)f0.x; r[1]=(_Float16)f0.y; r[2]=(_Float16)f0.z; r[3]=(_Float16)f0.w;
  r[4]=(_Float16)f1.x; r[5]=(_Float16)f1.y; r[6]=(_Float16)f1.z; r[7]=(_Float16)f1.w;
  r[8]=(_Float16)f2.x; r[9]=(_Float16)f2.y; r[10]=(_Float16)f2.z; r[11]=(_Float16)f2.w;
  r[12]=(_Float16)f3.x; r[13]=(_Float16)f3.y; r[14]=(_Float16)f3.z; r[15]=(_Float16)f3.w;
  return r;
}

// ---------------------------------------------------------------------------
// Implicit-GEMM conv, compile-time shape. Block = 4 waves = 64 consecutive
// output pixels of one row; ALL output-channel m-tiles handled in one block
// (MT accumulators) so the input halo tile is staged exactly once.
// A weights staged per K-chunk, full COUT width, pair-major [k/2][co][k&1]
// so each WMMA A-pair is one dword ds_load; B fragment reused across m-tiles.
// ---------------------------------------------------------------------------
template<int C0, int C1, int COUT, int KSZ, int STRIDE, int UPS, int RELU>
__global__ __launch_bounds__(128) void conv_wmma_t(
    const float* __restrict__ in0, int Hin0, int Win0,
    const float* __restrict__ in1,
    const float* __restrict__ w, const float* __restrict__ bias,
    float* __restrict__ out, int Hout, int Wout)
{
  constexpr int CIN = C0 + C1;
  constexpr int KK2 = KSZ * KSZ;
  constexpr int K   = CIN * KK2;
  constexpr int MT  = COUT / 16;
  constexpr int PAD = (KSZ == 3) ? (STRIDE == 1 ? 1 : 0) : 0; // XLA SAME pad_lo
  constexpr int BW  = 64;
  constexpr int IW  = (BW - 1) * STRIDE + KSZ;
  constexpr int BT_N = CIN * KSZ * IW;

  __shared__ _Float16 bT[BT_N];       // [ci][ky][ix_local]
  __shared__ _Float16 aT[32 * COUT];  // pair-major: [k_rel/2][co][k_rel&1]

  const int tid  = threadIdx.x;
  const int lane = tid & 31;
  const int wave = tid >> 5;
  const int lh   = lane >> 4;
  const int l16  = lane & 15;
  const int HWo  = Hout * Wout;
  const int p0   = blockIdx.x * BW;
  const int oy   = p0 / Wout;
  const int ox0  = p0 % Wout;
  const int Hl = UPS ? Hin0 * 2 : Hin0;
  const int Wl = UPS ? Win0 * 2 : Win0;

  // ---- stage input halo tile -> LDS (f16), zero-padded ----
  for (int idx = tid; idx < BT_N; idx += 128) {
    const int xi = idx % IW;
    const int t  = idx / IW;
    const int ky = t % KSZ;
    const int ci = t / KSZ;
    const int iy = oy  * STRIDE + ky - PAD;
    const int ix = ox0 * STRIDE + xi - PAD;
    float v = 0.f;
    if (iy >= 0 && iy < Hl && ix >= 0 && ix < Wl) {
      if (C1 == 0 || ci < C0) {
        const int y2 = UPS ? (iy >> 1) : iy;
        const int x2 = UPS ? (ix >> 1) : ix;
        v = in0[(ci * Hin0 + y2) * Win0 + x2];
      } else {
        v = in1[((ci - C0) * Hl + iy) * Wl + ix];
      }
    }
    bT[idx] = (_Float16)v;
  }

  const int xloc = wave * 16 + l16;

  v8f acc[MT];
  #pragma unroll
  for (int mt = 0; mt < MT; ++mt) acc[mt] = (v8f){};

  #pragma unroll
  for (int k0 = 0; k0 < K; k0 += 32) {
    // ---- cooperative full-width A-chunk stage (pair-major, zero-filled) ----
    __syncthreads();   // also covers bT on first iteration
    for (int i = tid; i < 32 * COUT; i += 128) {
      const int co   = i >> 5;
      const int krel = i & 31;
      const int kk = k0 + krel;
      aT[(krel >> 1) * (2 * COUT) + co * 2 + (krel & 1)] =
          (_Float16)((kk < K) ? w[co * K + kk] : 0.f);
    }
    __syncthreads();

    // ---- B fragment (shared across all m-tiles) ----
    v16h b;
    #pragma unroll
    for (int j = 0; j < 16; ++j) {
      const int kk = k0 + lh * 16 + j;   // constants after unroll
      _Float16 v = (_Float16)0.f;
      if (kk < K) {
        const int ci = kk / KK2, r = kk % KK2;
        const int ky = r / KSZ,  kx = r % KSZ;
        v = bT[(ci * KSZ + ky) * IW + xloc * STRIDE + kx];
      }
      b[j] = v;
    }

    #pragma unroll
    for (int mt = 0; mt < MT; ++mt) {
      v16h a;
      #pragma unroll
      for (int i = 0; i < 8; ++i) {
        const int kb = kbA(i, lh);     // even
        const h2 p = *(const h2*)&aT[(kb >> 1) * (2 * COUT) + (mt * 16 + l16) * 2];
        a[2*i]   = p[0];
        a[2*i+1] = p[1];
      }
      acc[mt] = wmma32(a, b, acc[mt]);
    }
  }

  const int p = p0 + xloc;
  #pragma unroll
  for (int mt = 0; mt < MT; ++mt) {
    #pragma unroll
    for (int r = 0; r < 8; ++r) {
      const int co = mt * 16 + r + lh * 8;
      float v = acc[mt][r] + bias[co];
      if (RELU) v = fmaxf(v, 0.f);
      out[co * HWo + p] = v;
    }
  }
}

// ---------------------------------------------------------------------------
__global__ void sp_pool_kernel(const float* __restrict__ emb,
                               const int* __restrict__ sp_flat,
                               const int* __restrict__ sp_seg,
                               float* __restrict__ sums, float* __restrict__ cnt)
{
  const int p = blockIdx.x * 256 + threadIdx.x;
  if (p >= P_PIX) return;
  const int idx = sp_flat[p];
  const int seg = sp_seg[p];
  atomicAdd(&cnt[seg], 1.0f);
  #pragma unroll
  for (int c = 0; c < C_EMB; ++c)
    atomicAdd(&sums[seg * C_EMB + c], emb[c * P_PIX + idx]);
}

__global__ void x_init_kernel(const float* __restrict__ sums,
                              const float* __restrict__ cnt,
                              float* __restrict__ x)
{
  const int i = blockIdx.x * 256 + threadIdx.x;
  if (i >= N_NODES * C_EMB) return;
  x[i] = sums[i] / fmaxf(cnt[i >> 4], 1.0f);
}

__global__ void deg_kernel(const int* __restrict__ dst, float* __restrict__ deg)
{
  const int e = blockIdx.x * 256 + threadIdx.x;
  if (e >= N_EDGES) return;
  atomicAdd(&deg[dst[e]], 1.0f);
}

// ---------------------------------------------------------------------------
// Node-conv edge MLP. Weights in LDS pair-major; activations edge-major
// [n][feature] so B reads are one 32B vector ds_load and D writes one
// 16B ds_store_b128 per lane.
// wL: layer0 (pad K=33->34) 544 | 5 hidden 256 each | out 256.
// ---------------------------------------------------------------------------
__global__ __launch_bounds__(128) void node_mlp_kernel(
    const float* __restrict__ x, const int* __restrict__ src,
    const int* __restrict__ dst, const float* __restrict__ angles,
    const float* __restrict__ win, const float* __restrict__ bin_,
    const float* __restrict__ wh, const float* __restrict__ bh,
    const float* __restrict__ wout, const float* __restrict__ bout,
    float* __restrict__ agg)
{
  __shared__ _Float16 wL[2080];
  __shared__ _Float16 act[4][16][16];   // [wave][edge n][feature]
  const int tid  = threadIdx.x;
  const int lane = tid & 31;
  const int wave = tid >> 5;
  const int lh   = lane >> 4;
  const int l16  = lane & 15;
  const int e_n  = (blockIdx.x * 4 + wave) * 16 + l16;
  const int s = src[e_n], d = dst[e_n];
  const float ang = angles[e_n];

  // ---- stage weights (pair-major [k/2][m][k&1]) ----
  for (int i = tid; i < 528; i += 128) {          // win (33x16)
    const int kk = i >> 4, m = i & 15;
    wL[(kk >> 1) * 32 + m * 2 + (kk & 1)] = (_Float16)win[i];
  }
  for (int i = tid; i < 16; i += 128)             // zero-pad kk=33
    wL[16 * 32 + i * 2 + 1] = (_Float16)0.f;
  for (int i = tid; i < 1280; i += 128) {         // wh (5x16x16)
    const int l = i >> 8, rr = i & 255, kk = rr >> 4, m = rr & 15;
    wL[544 + l * 256 + (kk >> 1) * 32 + m * 2 + (kk & 1)] = (_Float16)wh[i];
  }
  for (int i = tid; i < 256; i += 128) {          // wout (16x16)
    const int kk = i >> 4, m = i & 15;
    wL[1824 + (kk >> 1) * 32 + m * 2 + (kk & 1)] = (_Float16)wout[i];
  }
  __syncthreads();

  // ---- layer 0: K=33 padded to 64 (two WMMA chunks) ----
  v8f acc = {};
  #pragma unroll
  for (int k0 = 0; k0 < 64; k0 += 32) {
    v16h a;
    #pragma unroll
    for (int i = 0; i < 8; ++i) {
      const int kk = k0 + kbA(i, lh);
      h2 p = {};
      if (kk < 34) p = *(const h2*)&wL[(kk >> 1) * 32 + l16 * 2];
      a[2*i] = p[0]; a[2*i+1] = p[1];
    }
    v16h b = {};
    if (k0 == 0) {
      b = load16f(lh ? (x + d * 16) : (x + s * 16));
    } else if (lh == 0) {
      b[0] = (_Float16)ang;    // k=32 = angle, rest zero
    }
    acc = wmma32(a, b, acc);
  }
  {
    v8h pk;
    #pragma unroll
    for (int r = 0; r < 8; ++r)
      pk[r] = (_Float16)fmaxf(acc[r] + bin_[r + lh * 8], 0.f);
    *(v8h*)&act[wave][l16][lh * 8] = pk;
  }
  __syncthreads();

  // ---- 5 hidden + 1 output layer, K=16 padded to 32 ----
  for (int layer = 0; layer < 6; ++layer) {
    const int wbase = (layer < 5) ? (544 + layer * 256) : 1824;
    const float* B  = (layer < 5) ? (bh + layer * 16)   : bout;
    v16h a;
    #pragma unroll
    for (int i = 0; i < 8; ++i) {
      const int kk = kbA(i, lh);
      h2 p = {};
      if (kk < 16) p = *(const h2*)&wL[wbase + (kk >> 1) * 32 + l16 * 2];
      a[2*i] = p[0]; a[2*i+1] = p[1];
    }
    v16h b = {};
    if (lh == 0) b = *(const v16h*)&act[wave][l16][0];
    v8f c = {};
    c = wmma32(a, b, c);
    __syncthreads();
    if (layer < 5) {
      v8h pk;
      #pragma unroll
      for (int r = 0; r < 8; ++r)
        pk[r] = (_Float16)fmaxf(c[r] + B[r + lh * 8], 0.f);
      *(v8h*)&act[wave][l16][lh * 8] = pk;
      __syncthreads();
    } else {
      #pragma unroll
      for (int r = 0; r < 8; ++r) {
        const int m = r + lh * 8;
        atomicAdd(&agg[d * 16 + m], c[r] + B[m]);
      }
    }
  }
}

__global__ void node_update_kernel(const float* __restrict__ xin,
                                   const float* __restrict__ agg,
                                   const float* __restrict__ deg,
                                   float* __restrict__ xout)
{
  const int i = blockIdx.x * 256 + threadIdx.x;
  if (i >= N_NODES * C_EMB) return;
  const float dg = fmaxf(deg[i >> 4], 1.0f);
  xout[i] = fmaxf(xin[i] + agg[i] / dg, 0.f);
}

// ---------------------------------------------------------------------------
// Edge-conv MLP (48-wide, KIN = 32 or 80), both directions. Weights in LDS
// pair-major; activations edge-major ping-pong [n][48].
// ---------------------------------------------------------------------------
template<int KIN, int LEAKY>
__global__ __launch_bounds__(128) void edge_mlp_kernel(
    const float* __restrict__ x, const int* __restrict__ src,
    const int* __restrict__ dst, const float* __restrict__ e_in,
    const float* __restrict__ win, const float* __restrict__ bin_,
    const float* __restrict__ wh, const float* __restrict__ bh,
    const float* __restrict__ wout, const float* __restrict__ bout,
    float* __restrict__ e_out, float* __restrict__ sumsq)
{
  constexpr int W0    = KIN * C3_DIM;         // layer-0 region (KIN even)
  constexpr int WH    = 5 * C3_DIM * C3_DIM;
  constexpr int WOUTB = W0 + WH;
  __shared__ _Float16 wL[W0 + WH + C3_DIM * C3_DIM];
  __shared__ _Float16 act[4][2][16][C3_DIM];  // [wave][buf][edge n][feature]

  const int tid  = threadIdx.x;
  const int lane = tid & 31;
  const int wave = tid >> 5;
  const int lh   = lane >> 4;
  const int l16  = lane & 15;
  const int e_n  = (blockIdx.x * 4 + wave) * 16 + l16;
  const int s = src[e_n], d = dst[e_n];

  // ---- stage weights (pair-major [k/2][m][k&1], stride 96 halves) ----
  for (int i = tid; i < W0; i += 128) {
    const int kk = i / C3_DIM, m = i % C3_DIM;
    wL[(kk >> 1) * 96 + m * 2 + (kk & 1)] = (_Float16)win[i];
  }
  for (int i = tid; i < WH; i += 128) {
    const int l = i / (C3_DIM * C3_DIM), rr = i % (C3_DIM * C3_DIM);
    const int kk = rr / C3_DIM, m = rr % C3_DIM;
    wL[W0 + l * 2304 + (kk >> 1) * 96 + m * 2 + (kk & 1)] = (_Float16)wh[i];
  }
  for (int i = tid; i < C3_DIM * C3_DIM; i += 128) {
    const int kk = i / C3_DIM, m = i % C3_DIM;
    wL[WOUTB + (kk >> 1) * 96 + m * 2 + (kk & 1)] = (_Float16)wout[i];
  }
  __syncthreads();

  v8f res[2][3];

  for (int dir = 0; dir < 2; ++dir) {
    const int an = dir ? d : s;
    const int bn = dir ? s : d;
    int cur = 0;

    // ---- layer 0: KIN -> 48 ----
    #pragma unroll
    for (int mtile = 0; mtile < 3; ++mtile) {
      const int m2 = (mtile * 16 + l16) * 2;
      v8f acc = {};
      #pragma unroll
      for (int k0 = 0; k0 < KIN; k0 += 32) {
        v16h a;
        #pragma unroll
        for (int i = 0; i < 8; ++i) {
          const int kk = k0 + kbA(i, lh);
          h2 p = {};
          if (kk < KIN) p = *(const h2*)&wL[(kk >> 1) * 96 + m2];
          a[2*i] = p[0]; a[2*i+1] = p[1];
        }
        v16h b = {};
        if (k0 == 0) {
          b = load16f(lh ? (x + bn * 16) : (x + an * 16));
        } else if (k0 == 32) {
          b = load16f(e_in + e_n * C3_DIM + (lh ? 16 : 0));
        } else if (lh == 0) {
          b = load16f(e_in + e_n * C3_DIM + 32);
        }
        acc = wmma32(a, b, acc);
      }
      v8h pk;
      #pragma unroll
      for (int r = 0; r < 8; ++r)
        pk[r] = (_Float16)fmaxf(acc[r] + bin_[mtile * 16 + r + lh * 8], 0.f);
      *(v8h*)&act[wave][cur][l16][mtile * 16 + lh * 8] = pk;
    }
    __syncthreads();

    // ---- 5 hidden layers (48->48) + output layer ----
    for (int layer = 0; layer < 6; ++layer) {
      const int wbase = (layer < 5) ? (W0 + layer * 2304) : WOUTB;
      const float* B  = (layer < 5) ? (bh + layer * C3_DIM) : bout;
      #pragma unroll
      for (int mtile = 0; mtile < 3; ++mtile) {
        const int m2 = (mtile * 16 + l16) * 2;
        v8f acc = {};
        #pragma unroll
        for (int k0 = 0; k0 < 64; k0 += 32) {
          v16h a;
          #pragma unroll
          for (int i = 0; i < 8; ++i) {
            const int kk = k0 + kbA(i, lh);
            h2 p = {};
            if (kk < C3_DIM) p = *(const h2*)&wL[wbase + (kk >> 1) * 96 + m2];
            a[2*i] = p[0]; a[2*i+1] = p[1];
          }
          v16h b = {};
          if (k0 == 0)      b = *(const v16h*)&act[wave][cur][l16][lh * 16];
          else if (lh == 0) b = *(const v16h*)&act[wave][cur][l16][32];
          acc = wmma32(a, b, acc);
        }
        if (layer < 5) {
          v8h pk;
          #pragma unroll
          for (int r = 0; r < 8; ++r)
            pk[r] = (_Float16)fmaxf(acc[r] + B[mtile * 16 + r + lh * 8], 0.f);
          *(v8h*)&act[wave][1 - cur][l16][mtile * 16 + lh * 8] = pk;
        } else {
          #pragma unroll
          for (int r = 0; r < 8; ++r)
            acc[r] += B[mtile * 16 + (r + lh * 8)];
          res[dir][mtile] = acc;
        }
      }
      __syncthreads();
      cur ^= 1;
    }
  }

  // ---- combine directions: side loss + symmetric output ----
  float lsum = 0.f;
  #pragma unroll
  for (int mtile = 0; mtile < 3; ++mtile) {
    #pragma unroll
    for (int r = 0; r < 8; ++r) {
      const int m    = r + lh * 8;
      const float f  = res[0][mtile][r];
      const float bb = res[1][mtile][r];
      const float df = f - bb;
      lsum += df * df;
      float ev = 0.5f * (f + bb);
      if (LEAKY) ev = (ev > 0.f) ? ev : 0.01f * ev;
      e_out[e_n * C3_DIM + mtile * 16 + m] = ev;
    }
  }
  atomicAdd(sumsq, lsum);
}

__global__ void finalize_kernel(const float* __restrict__ sumsq,
                                float* __restrict__ out_scalar)
{
  const float denom = (float)N_EDGES * (float)C3_DIM;
  out_scalar[0] = 0.5f * (sumsq[0] / denom + sumsq[1] / denom);
}

// ---------------------------------------------------------------------------
extern "C" void kernel_launch(void* const* d_in, const int* in_sizes, int n_in,
                              void* d_out, int out_size, void* d_ws, size_t ws_size,
                              hipStream_t stream) {
  (void)in_sizes; (void)n_in; (void)out_size; (void)ws_size;
  const float* raw     = (const float*)d_in[0];
  const int*   ei      = (const int*)  d_in[1];
  const float* angles  = (const float*)d_in[2];
  const int*   sp_flat = (const int*)  d_in[4];
  const int*   sp_seg  = (const int*)  d_in[5];
  const float *w_e1=(const float*)d_in[6],  *b_e1=(const float*)d_in[7];
  const float *w_e2=(const float*)d_in[8],  *b_e2=(const float*)d_in[9];
  const float *w_b =(const float*)d_in[10], *b_b =(const float*)d_in[11];
  const float *w_d =(const float*)d_in[12], *b_d =(const float*)d_in[13];
  const float *w_o =(const float*)d_in[14], *b_o =(const float*)d_in[15];
  const float *nc1_win=(const float*)d_in[16], *nc1_bin=(const float*)d_in[17];
  const float *nc1_wh =(const float*)d_in[18], *nc1_bh =(const float*)d_in[19];
  const float *nc1_wout=(const float*)d_in[20],*nc1_bout=(const float*)d_in[21];
  const float *nc2_win=(const float*)d_in[22], *nc2_bin=(const float*)d_in[23];
  const float *nc2_wh =(const float*)d_in[24], *nc2_bh =(const float*)d_in[25];
  const float *nc2_wout=(const float*)d_in[26],*nc2_bout=(const float*)d_in[27];
  const float *ec1_win=(const float*)d_in[28], *ec2_win=(const float*)d_in[29];
  const float *ec1_bin=(const float*)d_in[30], *ec1_wh=(const float*)d_in[31];
  const float *ec1_bh =(const float*)d_in[32], *ec1_wout=(const float*)d_in[33];
  const float *ec1_bout=(const float*)d_in[34];
  const float *ec2_bin=(const float*)d_in[35], *ec2_wh=(const float*)d_in[36];
  const float *ec2_bh =(const float*)d_in[37], *ec2_wout=(const float*)d_in[38];
  const float *ec2_bout=(const float*)d_in[39];

  const int* src = ei;
  const int* dst = ei + N_EDGES;

  // ---- workspace layout (floats) ----
  float* ws  = (float*)d_ws;
  float* e1b = ws;                       // 32 * P
  float* e2b = e1b + 32 * P_PIX;         // 64 * P/4
  float* btb = e2b + 64 * (P_PIX / 4);   // 64 * P/4
  float* db  = btb + 64 * (P_PIX / 4);   // 32 * P
  // graph buffers overlaid on e2b (free after bottleneck conv)
  float* g = e2b;
  float* ebuf = g;  g += N_EDGES * C3_DIM;
  float* sums = g;  g += N_NODES * C_EMB;
  float* cnt  = g;  g += N_NODES;
  float* deg  = g;  g += N_NODES;
  float* xb   = g;  g += N_NODES * C_EMB;
  float* xb2  = g;  g += N_NODES * C_EMB;
  float* agg  = g;  g += N_NODES * C_EMB;
  float* sumsq= g;  g += 2;

  float* e_final = (float*)d_out;                   // [E, 48]
  float* emb     = e_final + N_EDGES * C3_DIM;      // [16, 1024, 1024]
  float* out_sc  = emb + C_EMB * P_PIX;             // scalar

  const dim3 blk(128);

  // ---- UNet (one block handles all output-channel tiles) ----
  conv_wmma_t<1, 0, 32, 3, 1, 0, 1><<<P_PIX/64, blk, 0, stream>>>(
      raw, H_IMG, H_IMG, nullptr, w_e1, b_e1, e1b, H_IMG, H_IMG);
  conv_wmma_t<32, 0, 64, 3, 2, 0, 1><<<(P_PIX/4)/64, blk, 0, stream>>>(
      e1b, H_IMG, H_IMG, nullptr, w_e2, b_e2, e2b, H_IMG/2, H_IMG/2);
  conv_wmma_t<64, 0, 64, 3, 1, 0, 1><<<(P_PIX/4)/64, blk, 0, stream>>>(
      e2b, H_IMG/2, H_IMG/2, nullptr, w_b, b_b, btb, H_IMG/2, H_IMG/2);
  conv_wmma_t<64, 32, 32, 3, 1, 1, 1><<<P_PIX/64, blk, 0, stream>>>(
      btb, H_IMG/2, H_IMG/2, e1b, w_d, b_d, db, H_IMG, H_IMG);
  conv_wmma_t<32, 0, 16, 1, 1, 0, 0><<<P_PIX/64, blk, 0, stream>>>(
      db, H_IMG, H_IMG, nullptr, w_o, b_o, emb, H_IMG, H_IMG);

  // ---- zero graph accumulators (sums..sumsq contiguous) ----
  hipMemsetAsync(sums, 0, (size_t)(g - sums) * sizeof(float), stream);

  // ---- superpixel mean pooling + degrees ----
  sp_pool_kernel<<<P_PIX/256, 256, 0, stream>>>(emb, sp_flat, sp_seg, sums, cnt);
  x_init_kernel<<<(N_NODES*C_EMB)/256, 256, 0, stream>>>(sums, cnt, xb);
  deg_kernel<<<N_EDGES/256, 256, 0, stream>>>(dst, deg);

  // ---- node conv 1 ----
  node_mlp_kernel<<<N_EDGES/64, blk, 0, stream>>>(
      xb, src, dst, angles, nc1_win, nc1_bin, nc1_wh, nc1_bh,
      nc1_wout, nc1_bout, agg);
  node_update_kernel<<<(N_NODES*C_EMB)/256, 256, 0, stream>>>(xb, agg, deg, xb2);

  // ---- edge conv 1 (no init edge feats) ----
  edge_mlp_kernel<32, 0><<<N_EDGES/64, blk, 0, stream>>>(
      xb2, src, dst, nullptr, ec1_win, ec1_bin, ec1_wh, ec1_bh,
      ec1_wout, ec1_bout, ebuf, &sumsq[0]);

  // ---- node conv 2 ----
  hipMemsetAsync(agg, 0, (size_t)(N_NODES*C_EMB) * sizeof(float), stream);
  node_mlp_kernel<<<N_EDGES/64, blk, 0, stream>>>(
      xb2, src, dst, angles, nc2_win, nc2_bin, nc2_wh, nc2_bh,
      nc2_wout, nc2_bout, agg);
  node_update_kernel<<<(N_NODES*C_EMB)/256, 256, 0, stream>>>(xb2, agg, deg, xb);

  // ---- edge conv 2 (with init edge feats) + final leaky relu ----
  edge_mlp_kernel<80, 1><<<N_EDGES/64, blk, 0, stream>>>(
      xb, src, dst, ebuf, ec2_win, ec2_bin, ec2_wh, ec2_bh,
      ec2_wout, ec2_bout, e_final, &sumsq[1]);

  // ---- side loss scalar ----
  finalize_kernel<<<1, 1, 0, stream>>>(sumsq, out_sc);
}
